// TetrahedralMessagePassing_73547019976728
// MI455X (gfx1250) — compile-verified, matrix operand order checked by reading, not run on hardware
//
#include <hip/hip_runtime.h>
#include <hip/hip_bf16.h>

// Problem constants (match reference)
#define BB 4
#define CC 256
#define NN 8192
#define KK 4
#define LL 3
#define MSGIN 640   // 2C + C/2
#define HID 512     // 2C
#define EHID 64     // C/4
#define EOUT 128    // C/2

typedef __attribute__((ext_vector_type(8)))  _Float16 v8h;
typedef __attribute__((ext_vector_type(16))) _Float16 v16h;
typedef __attribute__((ext_vector_type(8)))  float    v8f;

// Load one 16x32 f16 WMMA fragment (A-layout; B uses same pattern on the
// transposed weight matrix). p must point at row_base + kb + (lane>=16 ? 8:0).
__device__ __forceinline__ v16h load_frag(const _Float16* p) {
    v8h a = *(const v8h*)(p);
    v8h b = *(const v8h*)(p + 16);
    return __builtin_shufflevector(a, b, 0,1,2,3,4,5,6,7,8,9,10,11,12,13,14,15);
}

__device__ __forceinline__ v8f wmma_f16(v16h a, v16h b, v8f c) {
    return __builtin_amdgcn_wmma_f32_16x16x32_f16(false, a, false, b, (short)0, c, false, false);
}

// ---------------- pre-pass kernels ----------------

// feats[b][n][c] = x[b][c][n], f32 -> f16
__global__ void k_x_to_feats(const float* __restrict__ x, _Float16* __restrict__ feats) {
    int idx = blockIdx.x * blockDim.x + threadIdx.x;
    if (idx >= BB * NN * CC) return;
    int b = idx / (NN * CC);
    int rem = idx - b * (NN * CC);
    int n = rem / CC;
    int c = rem - n * CC;
    feats[idx] = (_Float16)x[((size_t)b * CC + c) * NN + n];
}

// dst[l][o][i] = (f16) src[l][i][o]   (weights transposed to out-major)
__global__ void k_convert_wT(const float* __restrict__ src, _Float16* __restrict__ dst,
                             int in_dim, int out_dim, int nmat) {
    int idx = blockIdx.x * blockDim.x + threadIdx.x;
    int per = in_dim * out_dim;
    if (idx >= nmat * per) return;
    int l = idx / per;
    int rem = idx - l * per;
    int o = rem / in_dim;
    int i = rem - o * in_dim;
    dst[idx] = (_Float16)src[(size_t)l * per + (size_t)i * out_dim + o];
}

// edge MLP: (N,K,3) -> (N,K,128), tiny, scalar f32
__global__ void k_edge(const float* __restrict__ points, const int* __restrict__ nbrs,
                       const float* __restrict__ W1, const float* __restrict__ b1,
                       const float* __restrict__ W2, const float* __restrict__ b2,
                       _Float16* __restrict__ edge16) {
    int idx = blockIdx.x * blockDim.x + threadIdx.x;
    if (idx >= NN * KK) return;
    int n = idx >> 2;
    int nb = nbrs[idx];
    float e0 = points[nb * 3 + 0] - points[n * 3 + 0];
    float e1 = points[nb * 3 + 1] - points[n * 3 + 1];
    float e2 = points[nb * 3 + 2] - points[n * 3 + 2];
    float h[EHID];
    for (int j = 0; j < EHID; ++j) {
        float v = e0 * W1[j] + e1 * W1[EHID + j] + e2 * W1[2 * EHID + j] + b1[j];
        h[j] = v > 0.0f ? v : 0.0f;
    }
    for (int c = 0; c < EOUT; ++c) {
        float s = b2[c];
        for (int j = 0; j < EHID; ++j) s += h[j] * W2[j * EOUT + c];
        edge16[(size_t)idx * EOUT + c] = (_Float16)s;
    }
}

// ---------------- msg kernel: 16 nodes x K=4 = 64 rows per block ----------------
// gathers [sender|receiver|edge] (64x640 f16) into LDS, two WMMA GEMM stages,
// in-register mean over K -> agg16 (f16)
__global__ __launch_bounds__(256) void k_msg(const _Float16* __restrict__ feats,
                       const int* __restrict__ nbrs,
                       const _Float16* __restrict__ edge16,
                       const _Float16* __restrict__ w1t, const float* __restrict__ b1,
                       const _Float16* __restrict__ w2t, const float* __restrict__ b2,
                       _Float16* __restrict__ agg16) {
    extern __shared__ char smem[];
    _Float16* sIn  = (_Float16*)smem;            // 64 x 640
    _Float16* sHid = sIn + 64 * MSGIN;           // 64 x 512

    const int tid = threadIdx.x;
    const int bI  = blockIdx.x / (NN / 16);
    const int tI  = blockIdx.x % (NN / 16);
    const int nodeBase = tI * 16;

    // gather: 64 rows x 80 chunks of 8 halves (16B)
    for (int ci = tid; ci < 64 * 80; ci += 256) {
        int r = ci / 80;
        int seg = ci - r * 80;
        int n = nodeBase + (r >> 2);
        int k = r & 3;
        const _Float16* src;
        if (seg < 32) {
            int nb = nbrs[n * KK + k];
            src = feats + ((size_t)bI * NN + nb) * CC + seg * 8;
        } else if (seg < 64) {
            src = feats + ((size_t)bI * NN + n) * CC + (seg - 32) * 8;
        } else {
            src = edge16 + ((size_t)(n * KK + k)) * EOUT + (seg - 64) * 8;
        }
        *(v8h*)(sIn + r * MSGIN + seg * 8) = *(const v8h*)src;
    }
    __syncthreads();

    const int lane = tid & 31;
    const int wave = tid >> 5;
    const int m    = lane & 15;
    const int hsel = (lane >> 4) * 8;  // fragment half select
    const int hi8  = (lane >> 4) * 8;  // C/D row offset

    // ---- stage 1: (64 x 640) @ (640 x 512) -> relu -> sHid ----
    // 4 col-tiles x 4 row-tiles of accumulators live at once:
    // per k-step: 4 A-frag ds loads + 4 B-frag global loads -> 16 WMMAs.
    {
        v8f acc[4][4] = {};   // [ct][rt]
        const _Float16* wrow[4];
#pragma unroll
        for (int ct = 0; ct < 4; ++ct)
            wrow[ct] = w1t + (size_t)(((wave + ct * 8) * 16) + m) * MSGIN + hsel;
        for (int kb = 0; kb < MSGIN; kb += 32) {
            v16h af[4];
#pragma unroll
            for (int rt = 0; rt < 4; ++rt)
                af[rt] = load_frag(sIn + (rt * 16 + m) * MSGIN + kb + hsel);
#pragma unroll
            for (int ct = 0; ct < 4; ++ct) {
                v16h bf = load_frag(wrow[ct] + kb);
#pragma unroll
                for (int rt = 0; rt < 4; ++rt)
                    acc[ct][rt] = wmma_f16(af[rt], bf, acc[ct][rt]);
            }
        }
#pragma unroll
        for (int ct = 0; ct < 4; ++ct) {
            int nb = (wave + ct * 8) * 16;
            float b1v = b1[nb + m];
#pragma unroll
            for (int rt = 0; rt < 4; ++rt)
#pragma unroll
                for (int v = 0; v < 8; ++v) {
                    float val = acc[ct][rt][v] + b1v;
                    val = val > 0.0f ? val : 0.0f;
                    sHid[(rt * 16 + hi8 + v) * HID + nb + m] = (_Float16)val;
                }
        }
    }
    __syncthreads();

    // ---- stage 2: (64 x 512) @ (512 x 256) -> mean over K -> agg ----
    {
        v8f acc[2][4] = {};
        const _Float16* wrow[2];
#pragma unroll
        for (int ct = 0; ct < 2; ++ct)
            wrow[ct] = w2t + (size_t)(((wave + ct * 8) * 16) + m) * HID + hsel;
        for (int kb = 0; kb < HID; kb += 32) {
            v16h af[4];
#pragma unroll
            for (int rt = 0; rt < 4; ++rt)
                af[rt] = load_frag(sHid + (rt * 16 + m) * HID + kb + hsel);
#pragma unroll
            for (int ct = 0; ct < 2; ++ct) {
                v16h bf = load_frag(wrow[ct] + kb);
#pragma unroll
                for (int rt = 0; rt < 4; ++rt)
                    acc[ct][rt] = wmma_f16(af[rt], bf, acc[ct][rt]);
            }
        }
#pragma unroll
        for (int ct = 0; ct < 2; ++ct) {
            int nb = (wave + ct * 8) * 16;
            float b2v = b2[nb + m];
#pragma unroll
            for (int rt = 0; rt < 4; ++rt) {
                // rows rt*16+hi8+{0..3} and {4..7} are full K-groups of one node each
                float s0 = acc[ct][rt][0] + acc[ct][rt][1] + acc[ct][rt][2] + acc[ct][rt][3];
                float s1 = acc[ct][rt][4] + acc[ct][rt][5] + acc[ct][rt][6] + acc[ct][rt][7];
                int nl = rt * 4 + (lane >> 4) * 2;
                agg16[((size_t)bI * NN + nodeBase + nl) * CC + nb + m]     = (_Float16)(s0 * 0.25f + b2v);
                agg16[((size_t)bI * NN + nodeBase + nl + 1) * CC + nb + m] = (_Float16)(s1 * 0.25f + b2v);
            }
        }
    }
}

// ---------------- upd kernel: 64 nodes per block ----------------
__global__ __launch_bounds__(256) void k_upd(const _Float16* __restrict__ feats,
                       const _Float16* __restrict__ agg16,
                       const _Float16* __restrict__ w1t, const float* __restrict__ b1,
                       const _Float16* __restrict__ w2t, const float* __restrict__ b2,
                       _Float16* __restrict__ featsNext, float* __restrict__ dout,
                       int isLast) {
    extern __shared__ char smem[];
    _Float16* sIn  = (_Float16*)smem;       // 64 x 512 : concat(feats, agg)
    _Float16* sHid = sIn + 64 * HID;        // 64 x 512

    const int tid = threadIdx.x;
    const int bI  = blockIdx.x / (NN / 64);
    const int tI  = blockIdx.x % (NN / 64);
    const int nodeBase = tI * 64;

    for (int ci = tid; ci < 64 * 64; ci += 256) {
        int r = ci >> 6;
        int seg = ci & 63;
        int n = nodeBase + r;
        const _Float16* src = (seg < 32)
            ? feats + ((size_t)bI * NN + n) * CC + seg * 8
            : agg16 + ((size_t)bI * NN + n) * CC + (seg - 32) * 8;
        *(v8h*)(sIn + r * HID + seg * 8) = *(const v8h*)src;
    }
    __syncthreads();

    const int lane = tid & 31;
    const int wave = tid >> 5;
    const int m    = lane & 15;
    const int hsel = (lane >> 4) * 8;
    const int hi8  = (lane >> 4) * 8;

    // ---- stage 1: (64 x 512) @ (512 x 512) -> relu -> sHid ----
    {
        v8f acc[4][4] = {};
        const _Float16* wrow[4];
#pragma unroll
        for (int ct = 0; ct < 4; ++ct)
            wrow[ct] = w1t + (size_t)(((wave + ct * 8) * 16) + m) * HID + hsel;
        for (int kb = 0; kb < HID; kb += 32) {
            v16h af[4];
#pragma unroll
            for (int rt = 0; rt < 4; ++rt)
                af[rt] = load_frag(sIn + (rt * 16 + m) * HID + kb + hsel);
#pragma unroll
            for (int ct = 0; ct < 4; ++ct) {
                v16h bf = load_frag(wrow[ct] + kb);
#pragma unroll
                for (int rt = 0; rt < 4; ++rt)
                    acc[ct][rt] = wmma_f16(af[rt], bf, acc[ct][rt]);
            }
        }
#pragma unroll
        for (int ct = 0; ct < 4; ++ct) {
            int nb = (wave + ct * 8) * 16;
            float b1v = b1[nb + m];
#pragma unroll
            for (int rt = 0; rt < 4; ++rt)
#pragma unroll
                for (int v = 0; v < 8; ++v) {
                    float val = acc[ct][rt][v] + b1v;
                    val = val > 0.0f ? val : 0.0f;
                    sHid[(rt * 16 + hi8 + v) * HID + nb + m] = (_Float16)val;
                }
        }
    }
    __syncthreads();

    // ---- stage 2: (64 x 512) @ (512 x 256) -> new feats (+ output) ----
    {
        v8f acc[2][4] = {};
        const _Float16* wrow[2];
#pragma unroll
        for (int ct = 0; ct < 2; ++ct)
            wrow[ct] = w2t + (size_t)(((wave + ct * 8) * 16) + m) * HID + hsel;
        for (int kb = 0; kb < HID; kb += 32) {
            v16h af[4];
#pragma unroll
            for (int rt = 0; rt < 4; ++rt)
                af[rt] = load_frag(sHid + (rt * 16 + m) * HID + kb + hsel);
#pragma unroll
            for (int ct = 0; ct < 2; ++ct) {
                v16h bf = load_frag(wrow[ct] + kb);
#pragma unroll
                for (int rt = 0; rt < 4; ++rt)
                    acc[ct][rt] = wmma_f16(af[rt], bf, acc[ct][rt]);
            }
        }
#pragma unroll
        for (int ct = 0; ct < 2; ++ct) {
            int nb = (wave + ct * 8) * 16;
            float b2v = b2[nb + m];
#pragma unroll
            for (int rt = 0; rt < 4; ++rt)
#pragma unroll
                for (int v = 0; v < 8; ++v) {
                    int row = rt * 16 + hi8 + v;
                    int n = nodeBase + row;
                    float val = acc[ct][rt][v] + b2v;
                    featsNext[((size_t)bI * NN + n) * CC + nb + m] = (_Float16)val;
                    if (isLast)
                        dout[((size_t)bI * CC + (nb + m)) * NN + n] = val;
                }
        }
    }
}

// ---------------- host launcher ----------------
extern "C" void kernel_launch(void* const* d_in, const int* in_sizes, int n_in,
                              void* d_out, int out_size, void* d_ws, size_t ws_size,
                              hipStream_t stream) {
    const float* x       = (const float*)d_in[0];
    const float* points  = (const float*)d_in[1];
    const int*   nbrs    = (const int*)d_in[2];
    const float* eW1     = (const float*)d_in[3];
    const float* eb1     = (const float*)d_in[4];
    const float* eW2     = (const float*)d_in[5];
    const float* eb2     = (const float*)d_in[6];
    const float* msgW1   = (const float*)d_in[7];
    const float* msgb1   = (const float*)d_in[8];
    const float* msgW2   = (const float*)d_in[9];
    const float* msgb2   = (const float*)d_in[10];
    const float* updW1   = (const float*)d_in[11];
    const float* updb1   = (const float*)d_in[12];
    const float* updW2   = (const float*)d_in[13];
    const float* updb2   = (const float*)d_in[14];
    float* out = (float*)d_out;

    // workspace layout (halves)
    _Float16* ws = (_Float16*)d_ws;
    const size_t FEAT = (size_t)BB * NN * CC;          // 8.39M halves
    _Float16* feats0 = ws;
    _Float16* feats1 = feats0 + FEAT;
    _Float16* agg    = feats1 + FEAT;
    _Float16* edge16 = agg + FEAT;                     // N*K*128
    _Float16* w1m    = edge16 + (size_t)NN * KK * EOUT;
    _Float16* w2m    = w1m + (size_t)LL * HID * MSGIN;
    _Float16* w1u    = w2m + (size_t)LL * CC * HID;
    _Float16* w2u    = w1u + (size_t)LL * HID * HID;

    const size_t SMEM_MSG = (size_t)64 * MSGIN * 2 + (size_t)64 * HID * 2; // 147456
    const size_t SMEM_UPD = (size_t)64 * HID * 2 * 2;                      // 131072
    (void)hipFuncSetAttribute((const void*)k_msg, hipFuncAttributeMaxDynamicSharedMemorySize, (int)SMEM_MSG);
    (void)hipFuncSetAttribute((const void*)k_upd, hipFuncAttributeMaxDynamicSharedMemorySize, (int)SMEM_UPD);

    // pre-passes
    {
        int tot = BB * NN * CC;
        k_x_to_feats<<<(tot + 255) / 256, 256, 0, stream>>>(x, feats0);
    }
    {
        int tot = LL * MSGIN * HID;
        k_convert_wT<<<(tot + 255) / 256, 256, 0, stream>>>(msgW1, w1m, MSGIN, HID, LL);
    }
    {
        int tot = LL * HID * CC;
        k_convert_wT<<<(tot + 255) / 256, 256, 0, stream>>>(msgW2, w2m, HID, CC, LL);
    }
    {
        int tot = LL * HID * HID;
        k_convert_wT<<<(tot + 255) / 256, 256, 0, stream>>>(updW1, w1u, HID, HID, LL);
    }
    {
        int tot = LL * HID * CC;
        k_convert_wT<<<(tot + 255) / 256, 256, 0, stream>>>(updW2, w2u, HID, CC, LL);
    }
    {
        int tot = NN * KK;
        k_edge<<<(tot + 255) / 256, 256, 0, stream>>>(points, nbrs, eW1, eb1, eW2, eb2, edge16);
    }

    _Float16* cur = feats0;
    _Float16* nxt = feats1;
    for (int l = 0; l < LL; ++l) {
        k_msg<<<BB * (NN / 16), 256, SMEM_MSG, stream>>>(
            cur, nbrs, edge16,
            w1m + (size_t)l * HID * MSGIN, msgb1 + (size_t)l * HID,
            w2m + (size_t)l * CC * HID,    msgb2 + (size_t)l * CC,
            agg);
        k_upd<<<BB * (NN / 64), 256, SMEM_UPD, stream>>>(
            cur, agg,
            w1u + (size_t)l * HID * HID, updb1 + (size_t)l * HID,
            w2u + (size_t)l * CC * HID,  updb2 + (size_t)l * CC,
            nxt, out, (l == LL - 1) ? 1 : 0);
        _Float16* t = cur; cur = nxt; nxt = t;
    }
}
